// convAttention_49048526520969
// MI455X (gfx1250) — matrix operand order repaired and checked by reading, not compiled, not run on hardware
//
#include <hip/hip_runtime.h>
#include <hip/hip_bf16.h>
#include <stdint.h>

// ---------------------------------------------------------------------------
// Problem constants
// ---------------------------------------------------------------------------
#define BATCH 8
#define CCH   384          // channels
#define HH    56
#define WW    56
#define HW    3136         // 56*56
#define NTOT  25088        // BATCH*HW  (== 196 * 128)
#define KDIM  384          // GEMM K (== 12 * 32)
#define HEADS 6
#define DPH   64

typedef __attribute__((ext_vector_type(16))) __bf16 v16bf;
typedef __attribute__((ext_vector_type(8)))  __bf16 v8bf;
typedef __attribute__((ext_vector_type(8)))  float  v8f;

#if defined(__gfx1250__) && __has_builtin(__builtin_amdgcn_tensor_load_to_lds) && \
    __has_builtin(__builtin_amdgcn_s_wait_tensorcnt)
#define USE_TDM 1
#else
#define USE_TDM 0
#endif

__device__ __forceinline__ float bf2f(unsigned short u) {
    union { unsigned int i; float f; } c; c.i = ((unsigned int)u) << 16; return c.f;
}
__device__ __forceinline__ unsigned short f2bf(float f) {
    union { float f; unsigned int i; } c; c.f = f;
    unsigned int u = c.i;
    u += 0x7FFFu + ((u >> 16) & 1u);         // round to nearest even
    return (unsigned short)(u >> 16);
}

// ---------------------------------------------------------------------------
// Conversion kernels (f32 -> bf16)
// ---------------------------------------------------------------------------
// x (B,C,H,W) f32  ->  xb [C][B*H*W] bf16   (c-major so it is the GEMM B operand)
__global__ void __launch_bounds__(256) cvt_x(const float* __restrict__ x,
                                             unsigned short* __restrict__ xb) {
    int t = blockIdx.x * 256 + threadIdx.x;
    if (t >= BATCH * CCH * HW) return;
    int hw = t % HW;
    int c  = (t / HW) % CCH;
    int b  = t / (HW * CCH);
    xb[c * NTOT + b * HW + hw] = f2bf(x[t]);
}

__global__ void __launch_bounds__(256) cvt_w(const float* __restrict__ s,
                                             unsigned short* __restrict__ d, int count) {
    int t = blockIdx.x * 256 + threadIdx.x;
    if (t < count) d[t] = f2bf(s[t]);
}

// ---------------------------------------------------------------------------
// TDM: one tensor_load_to_lds moves the whole 128x32 bf16 A tile into LDS,
// with hardware padding recreating the 80-byte LDS row pitch
// (pad_interval code 3 -> every 64B, pad_amount code 3 -> 4 DWORDs = 16B).
// Descriptor per CDNA5 ISA 08_async_tensor sec. 8.3/8.4.
// This toolchain exposes the 6-arg builtin:
//   (u32x4 g0, i32x8 g1, i32x4 g2, i32x4 g3, i32x8 extra, i32 cpol)
// ---------------------------------------------------------------------------
#if USE_TDM
typedef unsigned int u32x4 __attribute__((ext_vector_type(4)));
typedef int          i32x8 __attribute__((ext_vector_type(8)));
typedef int          i32x4 __attribute__((ext_vector_type(4)));

__device__ __forceinline__ unsigned lds_raw_addr(const void* p) {
    // Generic pointer to LDS: low 32 bits are the LDS byte offset (aperture form).
    return (unsigned)(unsigned long long)p;
}

__device__ __forceinline__ void tdm_load_tileA(const unsigned short* gsrc,
                                               unsigned short* lds_dst) {
    const unsigned long long ga = (unsigned long long)gsrc;
    u32x4 g0;
    g0.x = 1u;                                              // count=1 (valid user D#)
    g0.y = lds_raw_addr(lds_dst);                           // lds_addr
    g0.z = (unsigned)(ga & 0xFFFFFFFFull);                  // global_addr[31:0]
    g0.w = (unsigned)((ga >> 32) & 0x1FFFFFFull)            // global_addr[56:32]
         | 0x80000000u;                                     // type=2 ("image")
    i32x8 g1;
    g1[0] = (1 << 16)                                       // data_size = 2 bytes
          | (1 << 20)                                       // pad_enable
          | (3 << 22)                                       // pad_interval: 64B
          | (3 << 25);                                      // pad_amount: 16B
    g1[1] = (int)(((unsigned)KDIM & 0xFFFFu) << 16);        // tensor_dim0 = 384 (lo16)
    g1[2] = (int)(128u << 16);                              // dim0 hi=0 | tensor_dim1=128
    g1[3] = (int)(32u << 16);                               // dim1 hi=0 | tile_dim0=32
    g1[4] = 128;                                            // tile_dim1=128, tile_dim2=0
    g1[5] = KDIM;                                           // tensor_dim0_stride = 384
    g1[6] = 0;
    g1[7] = 0;
    const i32x4 z4 = {0, 0, 0, 0};                          // 2-D tile: groups 2/3 unused
    const i32x8 z8 = {0, 0, 0, 0, 0, 0, 0, 0};
    __builtin_amdgcn_tensor_load_to_lds(g0, g1, z4, z4, z8, 0);
}
#endif

// ---------------------------------------------------------------------------
// Tile staging: A via TDM (or fallback b128 copies), B with software transpose
// [K][N] global -> [n][k] LDS (padded rows of 40 ushorts = 80B).
// ---------------------------------------------------------------------------
__device__ __forceinline__ void stage_tiles(const unsigned short* __restrict__ A,
                                            const unsigned short* __restrict__ Bm,
                                            unsigned short* sAbuf,
                                            unsigned short* sBbuf,
                                            int m0, int n0, int k0, int tid) {
#if USE_TDM
    if (tid < 32) tdm_load_tileA(A + (m0 * KDIM + k0), sAbuf);
#else
#pragma unroll
    for (int it = 0; it < 2; ++it) {                 // A: 128x32 bf16 as 512 16B chunks
        int slot = tid + it * 256;
        int mm = slot >> 2, kc = slot & 3;
        const uint4 v = *(const uint4*)(A + (m0 + mm) * KDIM + k0 + kc * 8);
        *(uint4*)(&sAbuf[mm * 40 + kc * 8]) = v;
    }
#endif
#pragma unroll
    for (int it = 0; it < 8; ++it) {                 // B: 128 n * 16 k-pairs
        int slot = tid + it * 256;
        int nn = slot & 127, kp = slot >> 7;
        unsigned int lo = Bm[(k0 + 2 * kp)     * NTOT + n0 + nn];
        unsigned int hi = Bm[(k0 + 2 * kp + 1) * NTOT + n0 + nn];
        *(unsigned int*)(&sBbuf[nn * 40 + kp * 2]) = lo | (hi << 16);
    }
}

// ---------------------------------------------------------------------------
// WMMA GEMM:  D[M][NTOT] = A[M][384] * B[384][NTOT]   (bf16 in, f32 accumulate)
//   Block tile 128x128, BLK_K = 32, 256 threads = 8 waves, each wave 64Mx32N.
//   MODE 0: epilogue adds b_qkv[o], scales rows o<384 by 0.125 (q), stores bf16.
//   MODE 1: epilogue adds b_proj[o], scatters f32 into (B,C,H,W) output layout.
// ---------------------------------------------------------------------------
template <int MODE>
__device__ __forceinline__ void gemm_core(const unsigned short* __restrict__ A,
                                          const unsigned short* __restrict__ Bm,
                                          const float* __restrict__ bias,
                                          unsigned short* __restrict__ Dbf,
                                          float* __restrict__ Dfp) {
    __shared__ __align__(16) unsigned short sA[2][128 * 40];
    __shared__ __align__(16) unsigned short sB[2][128 * 40];

    const int tid  = threadIdx.x;
    const int lane = tid & 31;
    const int wid  = tid >> 5;
    const int wm   = wid >> 2;            // 0..1  -> 64 rows of M each
    const int wn   = wid & 3;             // 0..3  -> 32 cols of N each
    const int m0   = blockIdx.y * 128;
    const int n0   = blockIdx.x * 128;
    const int l15  = lane & 15;
    const int hsel = lane >> 4;

    v8f acc[4][2];
    const v8f zero = {0.f, 0.f, 0.f, 0.f, 0.f, 0.f, 0.f, 0.f};
#pragma unroll
    for (int mf = 0; mf < 4; ++mf)
#pragma unroll
        for (int nf = 0; nf < 2; ++nf) acc[mf][nf] = zero;

    const int NK = KDIM / 32;             // 12 k-steps

    stage_tiles(A, Bm, &sA[0][0], &sB[0][0], m0, n0, 0, tid);
#if USE_TDM
    __builtin_amdgcn_s_wait_tensorcnt(0);
#endif
    __syncthreads();

    int cur = 0;
    for (int ks = 0; ks < NK; ++ks) {
        if (ks + 1 < NK)
            stage_tiles(A, Bm, &sA[cur ^ 1][0], &sB[cur ^ 1][0],
                        m0, n0, (ks + 1) * 32, tid);

        // A frag (16-bit 16x32): lanes 0-15 carry K{0..7,16..23}, lanes 16-31 K{8..15,24..31}
        v16bf af[4];
#pragma unroll
        for (int mf = 0; mf < 4; ++mf) {
            int m  = wm * 64 + mf * 16 + l15;
            int kh = hsel * 8;
            const v8bf c0 = *(const v8bf*)(&sA[cur][m * 40 + kh]);
            const v8bf c1 = *(const v8bf*)(&sA[cur][m * 40 + kh + 16]);
            af[mf] = __builtin_shufflevector(c0, c1, 0, 1, 2, 3, 4, 5, 6, 7,
                                                     8, 9, 10, 11, 12, 13, 14, 15);
        }
        // B frag (16-bit 32x16): lanes 0-15 carry K{0..7,16..23}, lanes 16-31 K{8..15,24..31}
        v16bf bfr[2];
#pragma unroll
        for (int nf = 0; nf < 2; ++nf) {
            int nl = wn * 32 + nf * 16 + l15;
            int kk = hsel * 8;
            const v8bf c0 = *(const v8bf*)(&sB[cur][nl * 40 + kk]);
            const v8bf c1 = *(const v8bf*)(&sB[cur][nl * 40 + kk + 16]);
            bfr[nf] = __builtin_shufflevector(c0, c1, 0, 1, 2, 3, 4, 5, 6, 7,
                                                      8, 9, 10, 11, 12, 13, 14, 15);
        }
#pragma unroll
        for (int mf = 0; mf < 4; ++mf)
#pragma unroll
            for (int nf = 0; nf < 2; ++nf)
                acc[mf][nf] = __builtin_amdgcn_wmma_f32_16x16x32_bf16(
                    false, af[mf], false, bfr[nf], (short)0, acc[mf][nf], false, false);

#if USE_TDM
        __builtin_amdgcn_s_wait_tensorcnt(0);    // TDM writes done before barrier
#endif
        __syncthreads();
        cur ^= 1;
    }

    // ---- epilogue ----
#pragma unroll
    for (int mf = 0; mf < 4; ++mf)
#pragma unroll
        for (int nf = 0; nf < 2; ++nf)
#pragma unroll
            for (int e = 0; e < 8; ++e) {
                int o = m0 + wm * 64 + mf * 16 + e + hsel * 8;
                int n = n0 + wn * 32 + nf * 16 + l15;
                float v = acc[mf][nf][e] + bias[o];
                if (MODE == 0) {
                    if (o < CCH) v *= 0.125f;             // q scale = dph^-0.5
                    Dbf[o * NTOT + n] = f2bf(v);
                } else {
                    int b  = n / HW;
                    int hw = n - b * HW;
                    Dfp[(b * CCH + o) * HW + hw] = v;     // (B,C,H,W) layout
                }
            }
}

__global__ void __launch_bounds__(256) gemm_qkv(const unsigned short* A,
                                                const unsigned short* B,
                                                const float* bias,
                                                unsigned short* D) {
    gemm_core<0>(A, B, bias, D, nullptr);
}
__global__ void __launch_bounds__(256) gemm_proj(const unsigned short* A,
                                                 const unsigned short* B,
                                                 const float* bias,
                                                 float* D) {
    gemm_core<1>(A, B, bias, nullptr, D);
}

// ---------------------------------------------------------------------------
// Neighborhood attention: one thread per (head, pixel).
// qkvb: bf16 [1152][NTOT]  (q rows 0..383 already scaled, k 384..767, v 768..1151)
// Ob  : bf16 [384][NTOT]
// Out-of-bounds windows contribute score 0.0 (zero-padded k), and zero v.
// ---------------------------------------------------------------------------
__global__ void __launch_bounds__(256) attn(const unsigned short* __restrict__ qkvb,
                                            unsigned short* __restrict__ Ob) {
    int t = blockIdx.x * 256 + threadIdx.x;
    if (t >= HEADS * NTOT) return;
    int h = t / NTOT;
    int n = t - h * NTOT;
    int b = n / HW;
    int r = n - b * HW;
    int i = r / WW;
    int j = r - i * WW;

    const unsigned short* qbase = qkvb + (h * DPH) * NTOT + n;
    float q[DPH];
#pragma unroll
    for (int d = 0; d < DPH; ++d) q[d] = bf2f(qbase[d * NTOT]);

    float s[49];
    int   nn[49];
    const unsigned short* kbase = qkvb + (CCH + h * DPH) * NTOT;
    {
        int idx = 0;
#pragma unroll
        for (int dy = -3; dy <= 3; ++dy) {
            int ii = i + dy;
#pragma unroll
            for (int dx = -3; dx <= 3; ++dx, ++idx) {
                int jj = j + dx;
                bool ok = (ii >= 0) & (ii < HH) & (jj >= 0) & (jj < WW);
                int nnb = ok ? (n + dy * WW + dx) : n;   // safe address when padded
                nn[idx] = nnb;
                float acc = 0.f;
                if (ok) {
#pragma unroll
                    for (int d = 0; d < DPH; ++d)
                        acc += q[d] * bf2f(kbase[d * NTOT + nnb]);
                }
                s[idx] = ok ? acc : 0.f;                 // zero-padded k -> score 0
            }
        }
    }

    // softmax over the 49 window positions (all in registers)
    float mx = s[0];
#pragma unroll
    for (int a = 1; a < 49; ++a) mx = fmaxf(mx, s[a]);
    float sum = 0.f;
#pragma unroll
    for (int a = 0; a < 49; ++a) { s[a] = __expf(s[a] - mx); sum += s[a]; }
    float inv = 1.f / sum;
    {
        int idx = 0;
#pragma unroll
        for (int dy = -3; dy <= 3; ++dy) {
            int ii = i + dy;
#pragma unroll
            for (int dx = -3; dx <= 3; ++dx, ++idx) {
                int jj = j + dx;
                bool ok = (ii >= 0) & (ii < HH) & (jj >= 0) & (jj < WW);
                s[idx] = ok ? s[idx] * inv : 0.f;        // padded v contributes 0
            }
        }
    }

    const unsigned short* vbase = qkvb + (2 * CCH + h * DPH) * NTOT;
    unsigned short* obase = Ob + (h * DPH) * NTOT + n;
    for (int d = 0; d < DPH; ++d) {
        float o = 0.f;
#pragma unroll
        for (int a = 0; a < 49; ++a)
            o += s[a] * bf2f(vbase[d * NTOT + nn[a]]);
        obase[d * NTOT] = f2bf(o);
    }
}

// ---------------------------------------------------------------------------
// Launch
// ---------------------------------------------------------------------------
extern "C" void kernel_launch(void* const* d_in, const int* in_sizes, int n_in,
                              void* d_out, int out_size, void* d_ws, size_t ws_size,
                              hipStream_t stream) {
    const float* x      = (const float*)d_in[0];
    const float* w_qkv  = (const float*)d_in[1];
    const float* b_qkv  = (const float*)d_in[2];
    const float* w_proj = (const float*)d_in[3];
    const float* b_proj = (const float*)d_in[4];
    // d_in[5] = bias_table: multiplied by 0.0 in the reference -> unused.
    float* out = (float*)d_out;

    char* ws = (char*)d_ws;
    size_t off = 0;
    unsigned short* xb     = (unsigned short*)(ws + off); off += (size_t)CCH * NTOT * 2;      // 19.3 MB
    unsigned short* wqkvb  = (unsigned short*)(ws + off); off += (size_t)3 * CCH * KDIM * 2;  // 0.9 MB
    unsigned short* wprojb = (unsigned short*)(ws + off); off += (size_t)CCH * KDIM * 2;      // 0.3 MB
    unsigned short* qkvb   = (unsigned short*)(ws + off); off += (size_t)3 * CCH * NTOT * 2;  // 57.8 MB
    unsigned short* Ob     = (unsigned short*)(ws + off); off += (size_t)CCH * NTOT * 2;      // 19.3 MB

    cvt_x<<<(BATCH * CCH * HW + 255) / 256, 256, 0, stream>>>(x, xb);
    cvt_w<<<(3 * CCH * KDIM + 255) / 256, 256, 0, stream>>>(w_qkv, wqkvb, 3 * CCH * KDIM);
    cvt_w<<<(CCH * KDIM + 255) / 256, 256, 0, stream>>>(w_proj, wprojb, CCH * KDIM);

    gemm_qkv<<<dim3(NTOT / 128, (3 * CCH) / 128), 256, 0, stream>>>(wqkvb, xb, b_qkv, qkvb);
    attn<<<(HEADS * NTOT) / 256, 256, 0, stream>>>(qkvb, Ob);
    gemm_proj<<<dim3(NTOT / 128, CCH / 128), 256, 0, stream>>>(wprojb, Ob, b_proj, out);
}
